// CrossMutiHeadAttention_8761733284324
// MI455X (gfx1250) — matrix operand order repaired and checked by reading, not compiled, not run on hardware
//
#include <hip/hip_runtime.h>
#include <hip/hip_bf16.h>

#define BATCH 4
#define SQL 2048
#define SKV 2048
#define DMODEL 1024
#define DK 64
#define DV 128
#define NHEAD 8

typedef __attribute__((ext_vector_type(16))) __bf16       v16bf;
typedef __attribute__((ext_vector_type(8)))  float        v8f;
typedef __attribute__((ext_vector_type(4)))  unsigned int u32x4;

union Frag {
  v16bf bf;
  u32x4 u4[2];
};

__device__ __forceinline__ unsigned short f32_to_bf16(float f) {
  unsigned int x = __float_as_uint(f);
  unsigned int r = x + 0x7FFFu + ((x >> 16) & 1u);  // round-to-nearest-even
  return (unsigned short)(r >> 16);
}
__device__ __forceinline__ unsigned int pack_bf16x2(float lo, float hi) {
  return (unsigned int)f32_to_bf16(lo) | ((unsigned int)f32_to_bf16(hi) << 16);
}

// ---------------------------------------------------------------------------
// fp32 -> bf16 (grid-stride)
// ---------------------------------------------------------------------------
__global__ void cvt_f32_to_bf16(const float* __restrict__ src,
                                unsigned short* __restrict__ dst, int n) {
  int i = blockIdx.x * blockDim.x + threadIdx.x;
  int stride = gridDim.x * blockDim.x;
  for (; i < n; i += stride) dst[i] = f32_to_bf16(src[i]);
}

// ---------------------------------------------------------------------------
// fp32 [K][N] -> bf16 transposed [N][K] (one-time weight transpose)
// ---------------------------------------------------------------------------
__global__ void cvt_transpose_f32_to_bf16(const float* __restrict__ src,
                                          unsigned short* __restrict__ dst,
                                          int K, int N) {
  int i = blockIdx.x * blockDim.x + threadIdx.x;
  int n_total = K * N;
  int stride = gridDim.x * blockDim.x;
  for (; i < n_total; i += stride) {
    int k = i / N;
    int n = i - k * N;
    dst[(size_t)n * K + k] = f32_to_bf16(src[i]);
  }
}

// ---------------------------------------------------------------------------
// C = A[M,K] @ W[K,N]; A bf16 row-major, Wt = W pre-transposed bf16 [N][K].
// OUT_MODE: 0 = bf16 row-major (scaled), 1 = f32 row-major,
//           2 = bf16 written transposed as [B*H][dv][kv]  (for V projection)
// 256 threads = 8 waves; block tile 64x128; wave tile 32x32 (2x2 WMMA).
// ---------------------------------------------------------------------------
template <int OUT_MODE>
__global__ __launch_bounds__(256, 1) void gemm_bf16_wmma(
    const unsigned short* __restrict__ A, const unsigned short* __restrict__ Wt,
    void* __restrict__ C, int M, int N, int K, float outScale) {
  __shared__ __align__(16) unsigned short sA[64 * 32];   // [m][k]
  __shared__ __align__(16) unsigned short sB[128 * 32];  // [n][k]

  const int tid = threadIdx.x;
  const int lane = tid & 31;
  const int w = tid >> 5;
  const int g = (lane >> 4) & 1;
  const int ln = lane & 15;
  const int wr = w >> 2;  // 0..1
  const int wc = w & 3;   // 0..3
  const int blockM = blockIdx.x * 64;
  const int blockN = blockIdx.y * 128;

  v8f acc[2][2] = {};

  for (int k0 = 0; k0 < K; k0 += 32) {
    __syncthreads();
    // stage A: 64x32 (one 16B load per thread)
    {
      int e = tid * 8;
      int row = e >> 5;
      int col = e & 31;
      *(u32x4*)&sA[row * 32 + col] =
          *(const u32x4*)&A[(size_t)(blockM + row) * K + k0 + col];
    }
    // stage Wt: 128x32 (two 16B loads per thread, already [n][k])
    {
      int nn = tid >> 1;
      int cb = (tid & 1) * 16;
      const unsigned short* src = &Wt[(size_t)(blockN + nn) * K + k0 + cb];
      *(u32x4*)&sB[nn * 32 + cb] = *(const u32x4*)&src[0];
      *(u32x4*)&sB[nn * 32 + cb + 8] = *(const u32x4*)&src[8];
    }
    __syncthreads();

    Frag a[2], b[2];
#pragma unroll
    for (int mt = 0; mt < 2; ++mt) {
      int row = wr * 32 + mt * 16 + ln;
      a[mt].u4[0] = *(const u32x4*)&sA[row * 32 + 8 * g];
      a[mt].u4[1] = *(const u32x4*)&sA[row * 32 + 16 + 8 * g];
    }
#pragma unroll
    for (int nt = 0; nt < 2; ++nt) {
      int row = wc * 32 + nt * 16 + ln;
      b[nt].u4[0] = *(const u32x4*)&sB[row * 32 + 8 * g];
      b[nt].u4[1] = *(const u32x4*)&sB[row * 32 + 16 + 8 * g];
    }
#pragma unroll
    for (int mt = 0; mt < 2; ++mt)
#pragma unroll
      for (int nt = 0; nt < 2; ++nt)
        acc[mt][nt] = __builtin_amdgcn_wmma_f32_16x16x32_bf16(
            false, a[mt].bf, false, b[nt].bf, (short)0, acc[mt][nt], false,
            false);
  }

#pragma unroll
  for (int mt = 0; mt < 2; ++mt)
#pragma unroll
    for (int nt = 0; nt < 2; ++nt) {
      int n = blockN + wc * 32 + nt * 16 + ln;
      int mbase = blockM + wr * 32 + mt * 16 + 8 * g;
      if (OUT_MODE == 2) {
        // transposed V output: [b*H+h][dv][kv]; kv varies along r -> packed
        int h = n >> 7, dv = n & 127;
        int b_ = mbase >> 11, kv = mbase & 2047;
        u32x4 px;
#pragma unroll
        for (int j = 0; j < 4; ++j)
          px[j] = pack_bf16x2(acc[mt][nt][2 * j] * outScale,
                              acc[mt][nt][2 * j + 1] * outScale);
        *(u32x4*)&((unsigned short*)
                       C)[((size_t)(b_ * NHEAD + h) * DV + dv) * SKV + kv] = px;
      } else {
#pragma unroll
        for (int r = 0; r < 8; ++r) {
          float val = acc[mt][nt][r] * outScale;
          size_t idx = (size_t)(mbase + r) * N + n;
          if (OUT_MODE == 1)
            ((float*)C)[idx] = val;
          else
            ((unsigned short*)C)[idx] = f32_to_bf16(val);
        }
      }
    }
}

// ---------------------------------------------------------------------------
// Flash attention, fully transposed formulation.
//   S^T = K * Q^T   (A = K tile [kv][dk],  B = Q^T [dk][q])
//   O^T += V^T * P^T (A = V^T tile [dv][kv], B = P^T [kv][q])
// Per-lane softmax state (q index lives in the lane), 1 shuffle per reduction.
// Q:[B,SQ,H*DK] (pre-scaled by 1/sqrt(dk)), K:[B,SKV,H*DK], VT:[B*H][DV][SKV],
// O:[B,SQ,DMODEL], all bf16. 128 threads = 4 independent waves (16 q rows ea).
// grid = (SQ/64, B*H). No __syncthreads needed; K/V stream through L2.
// ---------------------------------------------------------------------------
__global__ __launch_bounds__(128, 1) void attn_flash_wmma(
    const unsigned short* __restrict__ Q, const unsigned short* __restrict__ Kv,
    const unsigned short* __restrict__ VT, unsigned short* __restrict__ O) {
  __shared__ __align__(16) unsigned short sP[4 * 16 * 32];  // per-wave P[q][kv]

  const int tid = threadIdx.x;
  const int lane = tid & 31;
  const int w = tid >> 5;  // 0..3
  const int g = (lane >> 4) & 1;
  const int ln = lane & 15;

  const int bh = blockIdx.y;
  const int b = bh >> 3;
  const int h = bh & 7;
  const int q0 = blockIdx.x * 64 + w * 16;

  // resident Q^T B-fragments (dk chunks of 32); lane column = q row (q0+ln)
  Frag qb[2];
  {
    const unsigned short* qrow =
        &Q[(size_t)(b * SQL + q0 + ln) * (NHEAD * DK) + h * DK];
#pragma unroll
    for (int c = 0; c < 2; ++c) {
      qb[c].u4[0] = *(const u32x4*)&qrow[c * 32 + 8 * g];
      qb[c].u4[1] = *(const u32x4*)&qrow[c * 32 + 16 + 8 * g];
    }
  }

  float mrow = -1e30f, lrow = 0.f;  // scalar per lane (q row = ln)
  v8f acc[8] = {};                  // O^T: 8 dv tiles
  unsigned short* pbuf = &sP[w * 16 * 32];

  const unsigned short* kbase = &Kv[(size_t)(b * SKV) * (NHEAD * DK) + h * DK];
  const unsigned short* vbase = &VT[(size_t)(b * NHEAD + h) * DV * SKV];

  for (int kv0 = 0; kv0 < SKV; kv0 += 32) {
    // ---- S^T tiles: s[nt][r] = S[q=ln][kv = kv0 + nt*16 + r + 8g] ----
    v8f s[2] = {};
#pragma unroll
    for (int nt = 0; nt < 2; ++nt) {
      const unsigned short* krow =
          &kbase[(size_t)(kv0 + nt * 16 + ln) * (NHEAD * DK)];
#pragma unroll
      for (int c = 0; c < 2; ++c) {
        Frag ka;
        ka.u4[0] = *(const u32x4*)&krow[c * 32 + 8 * g];
        ka.u4[1] = *(const u32x4*)&krow[c * 32 + 16 + 8 * g];
        s[nt] = __builtin_amdgcn_wmma_f32_16x16x32_bf16(
            false, ka.bf, false, qb[c].bf, (short)0, s[nt], false, false);
      }
    }

    // ---- online softmax: in-lane over 16 regs + ONE cross-half shuffle ----
    float tmax = fmaxf(s[0][0], s[1][0]);
#pragma unroll
    for (int r = 1; r < 8; ++r)
      tmax = fmaxf(tmax, fmaxf(s[0][r], s[1][r]));
    tmax = fmaxf(tmax, __shfl_xor(tmax, 16, 32));
    float mn = fmaxf(mrow, tmax);
    float corr = __expf(mrow - mn);
    mrow = mn;
    float rs = 0.f;
#pragma unroll
    for (int nt = 0; nt < 2; ++nt)
#pragma unroll
      for (int r = 0; r < 8; ++r) {
        float p = __expf(s[nt][r] - mn);
        s[nt][r] = p;
        rs += p;
      }
    rs += __shfl_xor(rs, 16, 32);
    lrow = lrow * corr + rs;
#pragma unroll
    for (int t = 0; t < 8; ++t)
#pragma unroll
      for (int r = 0; r < 8; ++r) acc[t][r] *= corr;  // scalar corr per lane

    // ---- P^T regs -> P[q][kv] in LDS (packed b128), reread as B-frag ----
#pragma unroll
    for (int nt = 0; nt < 2; ++nt) {
      u32x4 px;
#pragma unroll
      for (int j = 0; j < 4; ++j)
        px[j] = pack_bf16x2(s[nt][2 * j], s[nt][2 * j + 1]);
      *(u32x4*)&pbuf[ln * 32 + nt * 16 + 8 * g] = px;
    }
    Frag pb;  // same-wave LDS ops are in order; compiler inserts dscnt waits
    pb.u4[0] = *(const u32x4*)&pbuf[ln * 32 + 8 * g];
    pb.u4[1] = *(const u32x4*)&pbuf[ln * 32 + 16 + 8 * g];

    // ---- O^T += V^T * P^T : 8 dv tiles, V^T frags straight from L2 ----
#pragma unroll
    for (int t = 0; t < 8; ++t) {
      Frag va;
      const unsigned short* vrow = &vbase[(size_t)(t * 16 + ln) * SKV + kv0];
      va.u4[0] = *(const u32x4*)&vrow[8 * g];
      va.u4[1] = *(const u32x4*)&vrow[16 + 8 * g];
      acc[t] = __builtin_amdgcn_wmma_f32_16x16x32_bf16(
          false, va.bf, false, pb.bf, (short)0, acc[t], false, false);
    }
  }

  // ---- finalize: O^T regs -> ctx[b, q, h*128+dv]; dv varies along r ----
  float rdiv = 1.0f / lrow;
  unsigned short* orow = &O[(size_t)(b * SQL + q0 + ln) * DMODEL + h * DV];
#pragma unroll
  for (int t = 0; t < 8; ++t) {
    u32x4 px;
#pragma unroll
    for (int j = 0; j < 4; ++j)
      px[j] = pack_bf16x2(acc[t][2 * j] * rdiv, acc[t][2 * j + 1] * rdiv);
    *(u32x4*)&orow[t * 16 + 8 * g] = px;
  }
}

// ---------------------------------------------------------------------------
extern "C" void kernel_launch(void* const* d_in, const int* in_sizes, int n_in,
                              void* d_out, int out_size, void* d_ws,
                              size_t ws_size, hipStream_t stream) {
  (void)in_sizes;
  (void)n_in;
  (void)out_size;
  (void)ws_size;

  const float* enc = (const float*)d_in[0];  // [B,SKV,DMODEL]
  const float* pre = (const float*)d_in[1];  // [B,SQ,DMODEL]
  const float* Wq = (const float*)d_in[2];   // [DMODEL, H*DK]
  const float* Wk = (const float*)d_in[3];
  const float* Wv = (const float*)d_in[4];  // [DMODEL, DMODEL]
  const float* Wo = (const float*)d_in[5];

  const size_t N_ENC = (size_t)BATCH * SKV * DMODEL;
  const size_t N_PRE = (size_t)BATCH * SQL * DMODEL;
  const size_t N_WQ = (size_t)DMODEL * NHEAD * DK;
  const size_t N_WV = (size_t)DMODEL * DMODEL;
  const size_t N_Q = (size_t)BATCH * SQL * NHEAD * DK;
  const size_t N_V = (size_t)BATCH * SKV * DMODEL;
  const size_t N_CTX = (size_t)BATCH * SQL * DMODEL;

  unsigned char* ws = (unsigned char*)d_ws;
  size_t off = 0;
  auto take = [&](size_t elems) -> unsigned short* {
    unsigned short* p = (unsigned short*)(ws + off);
    off += ((elems * 2 + 255) & ~(size_t)255);
    return p;
  };
  unsigned short* enc_bf = take(N_ENC);
  unsigned short* pre_bf = take(N_PRE);
  unsigned short* WqT = take(N_WQ);  // [512][1024]
  unsigned short* WkT = take(N_WQ);
  unsigned short* WvT = take(N_WV);  // [1024][1024]
  unsigned short* WoT = take(N_WV);
  unsigned short* q_bf = take(N_Q);   // [B*SQ][512], pre-scaled
  unsigned short* k_bf = take(N_Q);   // [B*SKV][512]
  unsigned short* vT_bf = take(N_V);  // [B*H][128][2048]
  unsigned short* ctx_bf = take(N_CTX);

  auto cvt = [&](const float* s, unsigned short* d, size_t n) {
    cvt_f32_to_bf16<<<(int)((n + 2047) / 2048), 256, 0, stream>>>(s, d, (int)n);
  };
  auto cvtT = [&](const float* s, unsigned short* d, int K, int N) {
    int n = K * N;
    cvt_transpose_f32_to_bf16<<<(n + 255) / 256, 256, 0, stream>>>(s, d, K, N);
  };
  cvt(enc, enc_bf, N_ENC);
  cvt(pre, pre_bf, N_PRE);
  cvtT(Wq, WqT, DMODEL, NHEAD * DK);
  cvtT(Wk, WkT, DMODEL, NHEAD * DK);
  cvtT(Wv, WvT, DMODEL, DMODEL);
  cvtT(Wo, WoT, DMODEL, DMODEL);

  const int M = BATCH * SQL;  // 8192 (batch folded; weights shared)
  const float qscale = 0.125f;  // 1/sqrt(64) folded into q projection

  // q = (pre @ Wq) * scale   [8192,512] bf16
  gemm_bf16_wmma<0><<<dim3(M / 64, (NHEAD * DK) / 128), 256, 0, stream>>>(
      pre_bf, WqT, q_bf, M, NHEAD * DK, DMODEL, qscale);
  // k = enc @ Wk             [8192,512] bf16
  gemm_bf16_wmma<0><<<dim3(M / 64, (NHEAD * DK) / 128), 256, 0, stream>>>(
      enc_bf, WkT, k_bf, M, NHEAD * DK, DMODEL, 1.0f);
  // v = enc @ Wv, stored transposed as [B*H][dv][kv] bf16
  gemm_bf16_wmma<2><<<dim3(M / 64, DMODEL / 128), 256, 0, stream>>>(
      enc_bf, WvT, vT_bf, M, DMODEL, DMODEL, 1.0f);

  // flash attention -> ctx [8192,1024] bf16
  attn_flash_wmma<<<dim3(SQL / 64, BATCH * NHEAD), 128, 0, stream>>>(
      q_bf, k_bf, vT_bf, ctx_bf);

  // out = ctx @ Wo  (f32)
  gemm_bf16_wmma<1><<<dim3(M / 64, DMODEL / 128), 256, 0, stream>>>(
      ctx_bf, WoT, d_out, M, DMODEL, DMODEL, 1.0f);
}
